// StudentEncoder_20203526160388
// MI455X (gfx1250) — compile-verified, matrix operand order checked by reading, not run on hardware
//
#include <hip/hip_runtime.h>
#include <hip/hip_bf16.h>

// ---------------------------------------------------------------------------
// MI455X (gfx1250, wave32). All GEMM-shaped math on v_wmma_f32_16x16x32_bf16.
// All operands converted to bf16 exactly once (the whole working set is
// L2-resident on a 192MB-L2 part, so bf16 mirrors are nearly free and remove
// all per-tile v_cvt work from the WMMA inner loops).
// ---------------------------------------------------------------------------

typedef __attribute__((ext_vector_type(16))) __bf16 bf16x16;
typedef __attribute__((ext_vector_type(8)))  float  v8f;
typedef unsigned short bfraw;  // bf16 storage type at the host/kernel ABI

#define SEQ    2048
#define DMODEL 256

// 16-lane bf16 A/B fragment (32 K-elements across the wave halves) loaded as
// two raw 16-byte chunks. Layout per ISA 7.12.2 (16-bit A-matrix 16x32):
//   lanes 0-15  : K = k0+0..7   and k0+16..23
//   lanes 16-31 : K = k0+8..15  and k0+24..31
__device__ __forceinline__ bf16x16 load_frag_bf16(const __bf16* __restrict__ row,
                                                  int k0, int half) {
  union { bf16x16 v; uint4 q[2]; } f;
  f.q[0] = *reinterpret_cast<const uint4*>(row + k0 + half * 8);
  f.q[1] = *reinterpret_cast<const uint4*>(row + k0 + 16 + half * 8);
  return f.v;
}

// ---------------------------------------------------------------------------
// One-time f32 -> bf16 conversion (8 elements/thread, packed 16B store).
// ---------------------------------------------------------------------------
__global__ __launch_bounds__(256) void f32_to_bf16_kernel(
    const float* __restrict__ src, bfraw* __restrict__ dst, int n) {
  const int i = (blockIdx.x * 256 + threadIdx.x) * 8;
  if (i >= n) return;
  union { uint4 q; __bf16 h[8]; } u;
#pragma unroll
  for (int j = 0; j < 8; ++j) u.h[j] = (__bf16)src[i + j];
  *reinterpret_cast<uint4*>(dst + i) = u.q;
}

// ---------------------------------------------------------------------------
// Y[S x N] = act( X[S x K] @ W^T + bias ); X, W already bf16 (row-major,
// W is (N x K)). Writes fp32 output and/or a bf16 mirror. One wave computes
// a 16x64 tile (A fragment reused over 4 N tiles). grid=(S/16, N/64), block=32.
// ---------------------------------------------------------------------------
__global__ __launch_bounds__(32) void gemm_wmma_bf16(
    const bfraw* __restrict__ Xu, int lda,
    const bfraw* __restrict__ Wu, int ldw,
    const float* __restrict__ bias,
    float* __restrict__ Yf, bfraw* __restrict__ Yb, int ldy,
    int K, int act) {
  const __bf16* X = reinterpret_cast<const __bf16*>(Xu);
  const __bf16* W = reinterpret_cast<const __bf16*>(Wu);
  const int lane  = threadIdx.x & 31;
  const int m15   = lane & 15;
  const int half  = lane >> 4;
  const int mBase = blockIdx.x * 16;
  const int nBase = blockIdx.y * 64;
  const __bf16* xrow = X + (size_t)(mBase + m15) * lda;

  v8f acc[4] = {};
  for (int k0 = 0; k0 < K; k0 += 32) {
    if (k0 + 32 < K) __builtin_prefetch(xrow + k0 + 32, 0, 1);
    bf16x16 a = load_frag_bf16(xrow, k0, half);
#pragma unroll
    for (int t = 0; t < 4; ++t) {
      const __bf16* wrow = W + (size_t)(nBase + t * 16 + m15) * ldw;
      bf16x16 b = load_frag_bf16(wrow, k0, half);
      acc[t] = __builtin_amdgcn_wmma_f32_16x16x32_bf16(
          false, a, false, b, (short)0, acc[t], false, false);
    }
  }
  // C layout: VGPR i -> row mBase + half*8 + i ; column = tile + (lane&15)
#pragma unroll
  for (int t = 0; t < 4; ++t) {
    const int col = nBase + t * 16 + m15;
    const float bv = bias ? bias[col] : 0.0f;
#pragma unroll
    for (int i = 0; i < 8; ++i) {
      const int row = mBase + half * 8 + i;
      float v = acc[t][i] + bv;
      if (act) v = fmaxf(v, 0.0f);
      if (Yf) Yf[(size_t)row * ldy + col] = v;
      if (Yb) reinterpret_cast<__bf16*>(Yb)[(size_t)row * ldy + col] = (__bf16)v;
    }
  }
}

// ---------------------------------------------------------------------------
// Banded local attention: per (16-query block, head). hd=32, window +-32
// -> 80 keys. Scores: 5 WMMAs; in-register softmax with half-wave shuffle
// reductions; P staged in LDS *as bf16* (zero-padded K=96) so the second-pass
// A fragments are raw 16B ds loads; P@V: 3x2 WMMAs. Output written bf16.
// grid = (S/16, H=8), block = 32.
// ---------------------------------------------------------------------------
__global__ __launch_bounds__(32) void local_attn_wmma(
    const bfraw* __restrict__ Qu, const bfraw* __restrict__ Ku,
    const bfraw* __restrict__ Vu, bfraw* __restrict__ Ou, int S) {
  const __bf16* Q = reinterpret_cast<const __bf16*>(Qu);
  const __bf16* Kb = reinterpret_cast<const __bf16*>(Ku);
  const __bf16* V = reinterpret_cast<const __bf16*>(Vu);
  __bf16* O = reinterpret_cast<__bf16*>(Ou);

  __shared__ __align__(16) __bf16 P[16 * 96];
  const int lane = threadIdx.x & 31;
  const int m15  = lane & 15;
  const int half = lane >> 4;
  const int h    = blockIdx.y;
  const int qb   = blockIdx.x;
  const int qrow = qb * 16 + m15;

  // ---- scores = Q K^T / sqrt(hd) over 5 key tiles -------------------------
  bf16x16 aq = load_frag_bf16(Q + (size_t)qrow * DMODEL + h * 32, 0, half);
  v8f sc[5] = {};
#pragma unroll
  for (int t = 0; t < 5; ++t) {
    const int key = qb * 16 - 32 + t * 16 + m15;  // B column = key index
    bf16x16 bk;
    if (key >= 0 && key < S) {
      bk = load_frag_bf16(Kb + (size_t)key * DMODEL + h * 32, 0, half);
    } else {
#pragma unroll
      for (int i = 0; i < 16; ++i) bk[i] = (__bf16)0.0f;
    }
    sc[t] = __builtin_amdgcn_wmma_f32_16x16x32_bf16(
        false, aq, false, bk, (short)0, sc[t], false, false);
  }

  // ---- band mask + scale --------------------------------------------------
  const float scale = 0.17677669529663687f;  // 1/sqrt(32)
  float s[5][8];
#pragma unroll
  for (int t = 0; t < 5; ++t) {
#pragma unroll
    for (int i = 0; i < 8; ++i) {
      const int mq   = half * 8 + i;        // query row within tile
      const int krel = t * 16 + m15 - 32;   // key - query_block_base
      const int kAbs = qb * 16 + krel;
      const int d    = mq - krel;
      const bool ok  = (d <= 32) && (d >= -32) && (kAbs >= 0) && (kAbs < S);
      s[t][i] = ok ? sc[t][i] * scale : -1e9f;
    }
  }

  // ---- row softmax: reduce across the 16 lanes of each half-wave ----------
#pragma unroll
  for (int i = 0; i < 8; ++i) {
    float mx = -1e30f;
#pragma unroll
    for (int t = 0; t < 5; ++t) mx = fmaxf(mx, s[t][i]);
    for (int d = 1; d < 16; d <<= 1) mx = fmaxf(mx, __shfl_xor(mx, d, 32));
    float sum = 0.0f;
#pragma unroll
    for (int t = 0; t < 5; ++t) { s[t][i] = __expf(s[t][i] - mx); sum += s[t][i]; }
    for (int d = 1; d < 16; d <<= 1) sum += __shfl_xor(sum, d, 32);
    const float inv = 1.0f / sum;
#pragma unroll
    for (int t = 0; t < 5; ++t) s[t][i] *= inv;
  }

  // ---- transpose P (C layout -> A layout) through LDS as bf16, pad to 96 --
  for (int i = lane; i < 16 * 96; i += 32) P[i] = (__bf16)0.0f;
  __syncthreads();
#pragma unroll
  for (int t = 0; t < 5; ++t)
#pragma unroll
    for (int i = 0; i < 8; ++i)
      P[(half * 8 + i) * 96 + t * 16 + m15] = (__bf16)s[t][i];
  __syncthreads();

  // ---- out(16x32) = P(16x96) @ V(96x32) -----------------------------------
  v8f oacc[2] = {};
#pragma unroll
  for (int kb = 0; kb < 3; ++kb) {
    union { bf16x16 v; uint4 q[2]; } ap;
    ap.q[0] = *reinterpret_cast<const uint4*>(&P[m15 * 96 + kb * 32 + half * 8]);
    ap.q[1] = *reinterpret_cast<const uint4*>(&P[m15 * 96 + kb * 32 + 16 + half * 8]);
#pragma unroll
    for (int nt = 0; nt < 2; ++nt) {
      const int n = h * 32 + nt * 16 + m15;  // output channel (B column)
      bf16x16 bv;
#pragma unroll
      for (int i = 0; i < 8; ++i) {
        const int kr0 = kb * 32 + half * 8 + i;
        const int kr1 = kb * 32 + 16 + half * 8 + i;
        const int k0  = qb * 16 - 32 + kr0;
        const int k1  = qb * 16 - 32 + kr1;
        bv[i]     = (kr0 < 80 && k0 >= 0 && k0 < S) ? V[(size_t)k0 * DMODEL + n]
                                                    : (__bf16)0.0f;
        bv[i + 8] = (kr1 < 80 && k1 >= 0 && k1 < S) ? V[(size_t)k1 * DMODEL + n]
                                                    : (__bf16)0.0f;
      }
      oacc[nt] = __builtin_amdgcn_wmma_f32_16x16x32_bf16(
          false, ap.v, false, bv, (short)0, oacc[nt], false, false);
    }
  }
#pragma unroll
  for (int nt = 0; nt < 2; ++nt)
#pragma unroll
    for (int i = 0; i < 8; ++i)
      O[(size_t)(qb * 16 + half * 8 + i) * DMODEL + h * 32 + nt * 16 + m15] =
          (__bf16)oacc[nt][i];
}

// ---------------------------------------------------------------------------
// y = LayerNorm(x + a) * g + b, row length 256. One wave per row, 8 f32/lane,
// block = 256 -> 8 rows/block. Writes fp32 (next residual) + bf16 mirror
// (next GEMM operand) as one packed 16B store per lane.
// ---------------------------------------------------------------------------
__global__ __launch_bounds__(256) void resln_kernel(
    const float* __restrict__ x, const float* __restrict__ a,
    const float* __restrict__ g, const float* __restrict__ b,
    float* __restrict__ yf, bfraw* __restrict__ yb) {
  const int lane = threadIdx.x & 31;
  const int wid  = threadIdx.x >> 5;
  const int row  = blockIdx.x * 8 + wid;
  const float* xr = x + (size_t)row * DMODEL;
  const float* ar = a + (size_t)row * DMODEL;
  float v[8];
  float sum = 0.0f;
#pragma unroll
  for (int i = 0; i < 8; ++i) { v[i] = xr[lane * 8 + i] + ar[lane * 8 + i]; sum += v[i]; }
  for (int d = 1; d < 32; d <<= 1) sum += __shfl_xor(sum, d, 32);
  const float mean = sum * (1.0f / DMODEL);
  float var = 0.0f;
#pragma unroll
  for (int i = 0; i < 8; ++i) { const float t = v[i] - mean; var += t * t; }
  for (int d = 1; d < 32; d <<= 1) var += __shfl_xor(var, d, 32);
  const float inv = rsqrtf(var * (1.0f / DMODEL) + 1e-5f);
  union { uint4 q; __bf16 h[8]; } pk;
#pragma unroll
  for (int i = 0; i < 8; ++i) {
    const int c = lane * 8 + i;
    const float o = (v[i] - mean) * inv * g[c] + b[c];
    yf[(size_t)row * DMODEL + c] = o;
    pk.h[i] = (__bf16)o;
  }
  *reinterpret_cast<uint4*>(yb + (size_t)row * DMODEL + lane * 8) = pk.q;
}

// ---------------------------------------------------------------------------
// Hub integration, degenerate path (n_hubs == 1 -> softmax weights == 1):
//   v = hub[0]@Wv^T+bv ; att = v@Wo^T+bo ; cvec = att@Wint[:,256:]^T+bint.
// cvec becomes the bias of the final WMMA GEMM.
// ---------------------------------------------------------------------------
__global__ __launch_bounds__(256) void hub_kernel(
    const float* __restrict__ hub_emb, const float* __restrict__ Wi,
    const float* __restrict__ bi, const float* __restrict__ Wo,
    const float* __restrict__ bo, const float* __restrict__ Wint,
    const float* __restrict__ bint, float* __restrict__ cvec) {
  __shared__ float vvec[DMODEL];
  __shared__ float avec[DMODEL];
  const int n = threadIdx.x;
  float s = bi[2 * DMODEL + n];
  for (int k = 0; k < DMODEL; ++k)
    s += hub_emb[k] * Wi[(size_t)(2 * DMODEL + n) * DMODEL + k];
  vvec[n] = s;
  __syncthreads();
  float a = bo[n];
  for (int k = 0; k < DMODEL; ++k) a += vvec[k] * Wo[(size_t)n * DMODEL + k];
  avec[n] = a;
  __syncthreads();
  float c = bint[n];
  for (int k = 0; k < DMODEL; ++k)
    c += avec[k] * Wint[(size_t)n * 512 + DMODEL + k];
  cvec[n] = c;
}

// ---------------------------------------------------------------------------
// Output heads: 77 tiny dot products per row + passthrough copy of x.
// d_out layout: frame[S,6] torsion[S,4] var[S,1] dist[S,64] conf[S,1]
// tm[S,1] x[S,256].
// ---------------------------------------------------------------------------
__global__ __launch_bounds__(128) void heads_kernel(
    const float* __restrict__ x,
    const float* Wf, const float* bf, const float* Wt, const float* bt,
    const float* Wv, const float* bv, const float* Wd, const float* bd,
    const float* Wc, const float* bc, const float* Wm, const float* bm,
    float* __restrict__ out) {
  __shared__ float row[DMODEL];
  const int r = blockIdx.x;
  for (int i = threadIdx.x; i < DMODEL; i += blockDim.x)
    row[i] = x[(size_t)r * DMODEL + i];
  __syncthreads();
  const int j = threadIdx.x;
  const float* W = nullptr; float bias = 0.0f; size_t dst = 0;
  if (j < 6)        { W = Wf + j * DMODEL;        bias = bf[j];      dst = (size_t)r * 6 + j; }
  else if (j < 10)  { const int t = j - 6;  W = Wt + t * DMODEL; bias = bt[t]; dst = (size_t)SEQ * 6 + (size_t)r * 4 + t; }
  else if (j < 11)  { W = Wv;                     bias = bv[0];      dst = (size_t)SEQ * 10 + r; }
  else if (j < 75)  { const int t = j - 11; W = Wd + t * DMODEL; bias = bd[t]; dst = (size_t)SEQ * 11 + (size_t)r * 64 + t; }
  else if (j < 76)  { W = Wc;                     bias = bc[0];      dst = (size_t)SEQ * 75 + r; }
  else if (j < 77)  { W = Wm;                     bias = bm[0];      dst = (size_t)SEQ * 76 + r; }
  if (j < 77) {
    float s = bias;
    for (int k = 0; k < DMODEL; ++k) s += row[k] * W[k];
    out[dst] = s;
  }
  for (int i = threadIdx.x; i < DMODEL; i += blockDim.x)
    out[(size_t)SEQ * 77 + (size_t)r * DMODEL + i] = row[i];
}

// ---------------------------------------------------------------------------
// Host orchestration.
// ---------------------------------------------------------------------------
extern "C" void kernel_launch(void* const* d_in, const int* in_sizes, int n_in,
                              void* d_out, int out_size, void* d_ws, size_t ws_size,
                              hipStream_t stream) {
  (void)in_sizes; (void)n_in; (void)out_size; (void)ws_size;
  const int S = SEQ;
  int ix = 0;
  auto nxt = [&]() { return (const float*)d_in[ix++]; };

  const float* emb = nxt();                            // embeddings (2048,512)
  const float* inW = nxt(); const float* inB = nxt();  // in_emb
  struct Layer { const float *qW,*qb,*kW,*kb,*vW,*vb,*oW,*ob,*g1,*be1,*g2,*be2,*f1W,*f1b,*f2W,*f2b; } L[3];
  for (int l = 0; l < 3; ++l) {
    L[l].qW=nxt(); L[l].qb=nxt(); L[l].kW=nxt(); L[l].kb=nxt();
    L[l].vW=nxt(); L[l].vb=nxt(); L[l].oW=nxt(); L[l].ob=nxt();
    L[l].g1=nxt(); L[l].be1=nxt(); L[l].g2=nxt(); L[l].be2=nxt();
    L[l].f1W=nxt(); L[l].f1b=nxt(); L[l].f2W=nxt(); L[l].f2b=nxt();
  }
  struct Se3 { const float *l1W,*l1b,*l2W,*l2b,*g,*be; } Z[2];
  for (int z = 0; z < 2; ++z) {
    Z[z].l1W=nxt(); Z[z].l1b=nxt(); Z[z].l2W=nxt(); Z[z].l2b=nxt();
    Z[z].g=nxt();   Z[z].be=nxt();
  }
  const float* hubEmb=nxt(); const float* hubWi=nxt(); const float* hubBi=nxt();
  const float* hubOW=nxt();  const float* hubOb=nxt();
  const float* hubIW=nxt();  const float* hubIb=nxt();
  const float* hWf=nxt(); const float* hbf=nxt();
  const float* hWt=nxt(); const float* hbt=nxt();
  const float* hWv=nxt(); const float* hbv=nxt();
  const float* hWd=nxt(); const float* hbd=nxt();
  const float* hWc=nxt(); const float* hbc=nxt();
  const float* hWm=nxt(); const float* hbm=nxt();

  // ---- workspace carve-up -------------------------------------------------
  // fp32: 3x (S,256) buffers + cvec ; then a bf16 arena.
  float* bufA = (float*)d_ws;                   // x (fp32)
  float* bufB = bufA + (size_t)S * DMODEL;      // temp x (fp32)
  float* bufP = bufB + (size_t)S * DMODEL;      // projection / MLP out (fp32)
  float* cvec = bufP + (size_t)S * DMODEL;      // (256)
  bfraw* bfp  = (bfraw*)(cvec + DMODEL);
  auto abf = [&](size_t n) { bfraw* p = bfp; bfp += n; return p; };

  bfraw* embB = abf((size_t)S * 512);           // bf16 embeddings
  bfraw* wIn  = abf(256 * 512);
  bfraw *wq[3], *wk[3], *wv[3], *wo[3], *wf1[3], *wf2[3];
  for (int l = 0; l < 3; ++l) {
    wq[l] = abf(256 * 256); wk[l] = abf(256 * 256);
    wv[l] = abf(256 * 256); wo[l] = abf(256 * 256);
    wf1[l] = abf(1024 * 256); wf2[l] = abf(256 * 1024);
  }
  bfraw *wl1[2], *wl2[2];
  for (int z = 0; z < 2; ++z) { wl1[z] = abf(256 * 256); wl2[z] = abf(256 * 256); }
  bfraw* wHubI = abf(256 * 512);
  bfraw* xbfA  = abf((size_t)S * DMODEL);       // bf16 mirror of x
  bfraw* xbfB  = abf((size_t)S * DMODEL);
  bfraw* qb    = abf((size_t)S * DMODEL);
  bfraw* kb    = abf((size_t)S * DMODEL);
  bfraw* vb    = abf((size_t)S * DMODEL);
  bfraw* attb  = abf((size_t)S * DMODEL);
  bfraw* hb    = abf((size_t)S * 1024);

  auto CVT = [&](const float* src, bfraw* dst, int n) {
    f32_to_bf16_kernel<<<dim3((n / 8 + 255) / 256), dim3(256), 0, stream>>>(src, dst, n);
  };
  auto GEMM = [&](const bfraw* X, int lda, const bfraw* W, int ldw,
                  const float* bias, float* Yf, bfraw* Yb, int ldy,
                  int K, int N, int act) {
    dim3 grid(S / 16, N / 64);
    gemm_wmma_bf16<<<grid, dim3(32), 0, stream>>>(X, lda, W, ldw, bias, Yf, Yb, ldy, K, act);
  };
  auto RESLN = [&](const float* x, const float* a, const float* g,
                   const float* b, float* yf, bfraw* yb) {
    resln_kernel<<<dim3(S / 8), dim3(256), 0, stream>>>(x, a, g, b, yf, yb);
  };

  // ---- one-time bf16 conversions ------------------------------------------
  CVT(emb, embB, S * 512);
  CVT(inW, wIn, 256 * 512);
  for (int l = 0; l < 3; ++l) {
    CVT(L[l].qW, wq[l], 256 * 256); CVT(L[l].kW, wk[l], 256 * 256);
    CVT(L[l].vW, wv[l], 256 * 256); CVT(L[l].oW, wo[l], 256 * 256);
    CVT(L[l].f1W, wf1[l], 1024 * 256); CVT(L[l].f2W, wf2[l], 256 * 1024);
  }
  for (int z = 0; z < 2; ++z) {
    CVT(Z[z].l1W, wl1[z], 256 * 256); CVT(Z[z].l2W, wl2[z], 256 * 256);
  }
  CVT(hubIW, wHubI, 256 * 512);

  // ---- input embedding (fp32 x + bf16 mirror) -----------------------------
  GEMM(embB, 512, wIn, 512, inB, bufA, xbfA, DMODEL, 512, DMODEL, 0);

  // ---- 3 local-attention blocks ------------------------------------------
  for (int l = 0; l < 3; ++l) {
    GEMM(xbfA, DMODEL, wq[l], DMODEL, L[l].qb, nullptr, qb, DMODEL, DMODEL, DMODEL, 0);
    GEMM(xbfA, DMODEL, wk[l], DMODEL, L[l].kb, nullptr, kb, DMODEL, DMODEL, DMODEL, 0);
    GEMM(xbfA, DMODEL, wv[l], DMODEL, L[l].vb, nullptr, vb, DMODEL, DMODEL, DMODEL, 0);
    local_attn_wmma<<<dim3(S / 16, 8), dim3(32), 0, stream>>>(qb, kb, vb, attb, S);
    GEMM(attb, DMODEL, wo[l], DMODEL, L[l].ob, bufP, nullptr, DMODEL, DMODEL, DMODEL, 0);
    RESLN(bufA, bufP, L[l].g1, L[l].be1, bufB, xbfB);
    GEMM(xbfB, DMODEL, wf1[l], DMODEL, L[l].f1b, nullptr, hb, 1024, DMODEL, 1024, 1);
    GEMM(hb, 1024, wf2[l], 1024, L[l].f2b, bufP, nullptr, DMODEL, 1024, DMODEL, 0);
    RESLN(bufB, bufP, L[l].g2, L[l].be2, bufA, xbfA);
  }

  // ---- 2 SE3 MLP blocks ---------------------------------------------------
  float* xF = bufA; bfraw* xB = xbfA;
  float* tF = bufB; bfraw* tB = xbfB;
  for (int z = 0; z < 2; ++z) {
    GEMM(xB, DMODEL, wl1[z], DMODEL, Z[z].l1b, nullptr, qb, DMODEL, DMODEL, DMODEL, 1);
    GEMM(qb, DMODEL, wl2[z], DMODEL, Z[z].l2b, bufP, nullptr, DMODEL, DMODEL, DMODEL, 0);
    RESLN(xF, bufP, Z[z].g, Z[z].be, tF, tB);
    float* sf = xF; xF = tF; tF = sf;
    bfraw* sb = xB; xB = tB; tB = sb;
  }

  // ---- hub integration (degenerate single-hub path) -----------------------
  hub_kernel<<<dim3(1), dim3(256), 0, stream>>>(hubEmb, hubWi, hubBi, hubOW,
                                                hubOb, hubIW, hubIb, cvec);
  // out = x @ Wint[:, :256]^T + cvec   (cvec already holds att-part + bint)
  GEMM(xB, DMODEL, wHubI, 512, cvec, tF, nullptr, DMODEL, DMODEL, DMODEL, 0);

  // ---- heads + passthrough ------------------------------------------------
  heads_kernel<<<dim3(S), dim3(128), 0, stream>>>(
      tF, hWf, hbf, hWt, hbt, hWv, hbv, hWd, hbd, hWc, hbc, hWm, hbm,
      (float*)d_out);
}
